// LSTM_42064909697157
// MI455X (gfx1250) — compile-verified, matrix operand order checked by reading, not run on hardware
//
#include <hip/hip_runtime.h>

// ---------------------------------------------------------------------------
// Problem constants (match reference)
// ---------------------------------------------------------------------------
constexpr int T    = 512;
constexpr int E    = 128;
constexpr int NN   = 32;    // nodes
constexpr int HH   = 256;   // hidden
constexpr int G4   = 1024;  // 4*H
constexpr int DIN0 = 64;
constexpr int DIN1 = 512;   // 2*H

typedef __bf16 bf16;
typedef __attribute__((ext_vector_type(16))) __bf16 v16bf;
typedef __attribute__((ext_vector_type(8)))  __bf16 v8bf;
typedef __attribute__((ext_vector_type(8)))  float  v8f;

__device__ __forceinline__ float sigmoidf(float x) {
    return 1.0f / (1.0f + __expf(-x));
}

// Load one 16x32 bf16 WMMA operand fragment from a row-major matrix.
// Works for A (rows = M tile) and for B when B[k][n] = W[n][k] with W
// row-major [N][K] (rows = N tile). Per ISA 7.12.2:
//   lane<16 : row = lane,    elems 0..7 -> K k0+0..7,  elems 8..15 -> K k0+16..23
//   lane>=16: row = lane-16, elems 0..7 -> K k0+8..15, elems 8..15 -> K k0+24..31
__device__ __forceinline__ v16bf load_frag16(const bf16* p, int ld, int lane) {
    int r  = lane & 15;
    int kh = (lane >> 4) * 8;
    const bf16* base = p + (long)r * ld + kh;
    v8bf lo = *(const v8bf*)(base);        // 16B load
    v8bf hi = *(const v8bf*)(base + 16);   // 16B load
    return __builtin_shufflevector(lo, hi, 0,1,2,3,4,5,6,7,8,9,10,11,12,13,14,15);
}

// ---------------------------------------------------------------------------
// f32 -> bf16 conversion (grid-stride)
// ---------------------------------------------------------------------------
__global__ void cvt_bf16_kernel(const float* __restrict__ in,
                                bf16* __restrict__ out, long n) {
    long i = (long)blockIdx.x * blockDim.x + threadIdx.x;
    long stride = (long)gridDim.x * blockDim.x;
    for (; i < n; i += stride) out[i] = (bf16)in[i];
}

// ---------------------------------------------------------------------------
// Zero-init node state
// ---------------------------------------------------------------------------
__global__ void init_state_kernel(bf16* __restrict__ hn, float* __restrict__ cn) {
    int i = blockIdx.x * blockDim.x + threadIdx.x;
    if (i < NN * HH) { hn[i] = (bf16)0.0f; cn[i] = 0.0f; }
}

// ---------------------------------------------------------------------------
// Input projection GEMM:  G[m][n] = sum_k X[m][k]*W[n][k] + b[n]
// X: [M,K] bf16 row-major, W: [N,K] bf16 row-major, G: [M,N] f32.
// One wave computes a 16x64 tile (4 WMMA accumulators, A reused 4x).
// Fragments are software-pipelined: iteration k's WMMAs issue after
// iteration k+1's loads, so s_wait_loadcnt can tolerate nonzero counts
// instead of stalling on a full round-trip per WMMA.
// ---------------------------------------------------------------------------
__global__ void proj_gemm_kernel(const bf16* __restrict__ X,
                                 const bf16* __restrict__ W,
                                 const float* __restrict__ bias,
                                 float* __restrict__ G,
                                 int M, int N, int K) {
    int lane = threadIdx.x & 31;
    int wave = threadIdx.x >> 5;
    int job  = blockIdx.x * (blockDim.x >> 5) + wave;
    int tilesN = N / 64;
    int mtile  = job / tilesN;
    int ngrp   = job % tilesN;
    if (mtile * 16 >= M) return;

    const int m0 = mtile * 16;
    const int n0 = ngrp * 64;

    const bf16* Abase = X + (long)m0 * K;
    const bf16* Bbase[4];
#pragma unroll
    for (int j = 0; j < 4; ++j) Bbase[j] = W + (long)(n0 + 16 * j) * K;

    v8f acc[4] = {};

    // Prologue: load fragments for k0 = 0.
    v16bf a_cur = load_frag16(Abase, K, lane);
    v16bf b_cur[4];
#pragma unroll
    for (int j = 0; j < 4; ++j) b_cur[j] = load_frag16(Bbase[j], K, lane);

    for (int k0 = 0; k0 < K; k0 += 32) {
        v16bf a_nxt;
        v16bf b_nxt[4];
        const int k1 = k0 + 32;
        if (k1 < K) {
            // Issue next iteration's loads before consuming current frags.
            a_nxt = load_frag16(Abase + k1, K, lane);
#pragma unroll
            for (int j = 0; j < 4; ++j)
                b_nxt[j] = load_frag16(Bbase[j] + k1, K, lane);
        }
#pragma unroll
        for (int j = 0; j < 4; ++j) {
            acc[j] = __builtin_amdgcn_wmma_f32_16x16x32_bf16(
                false, a_cur, false, b_cur[j], (short)0, acc[j], false, false);
        }
        if (k1 < K) {
            a_cur = a_nxt;
#pragma unroll
            for (int j = 0; j < 4; ++j) b_cur[j] = b_nxt[j];
        }
    }

    int mrow = m0 + (lane >> 4) * 8;
#pragma unroll
    for (int j = 0; j < 4; ++j) {
        int n = n0 + 16 * j + (lane & 15);
        float bv = bias[n];
#pragma unroll
        for (int v = 0; v < 8; ++v)
            G[(long)(mrow + v) * N + n] = acc[j][v] + bv;
    }
}

// ---------------------------------------------------------------------------
// One LSTM scan step for one direction.  Single workgroup (1024 thr = 32
// waves) so the cross-step recurrence is ordered by kernel launch order.
//  1) Hproj[32][1024] = hn(bf16) @ Whh^T  via WMMA, into 128KB dynamic LDS
//  2) deterministic per-node edge lists (no float atomics)
//  3) per-(node,h) thread: full LSTM cell per owned edge, per-edge h output,
//     fixed-order segment-mean -> new node state (hn bf16, cn f32)
// ---------------------------------------------------------------------------
__global__ void lstm_step_kernel(const float* __restrict__ Gin,   // [T,E,4H]
                                 const bf16*  __restrict__ Whh,   // [4H,H]
                                 const int*   __restrict__ src,   // [T,E]
                                 const int*   __restrict__ dst,   // [T,E]
                                 bf16*  __restrict__ hn,          // [NN,HH]
                                 float* __restrict__ cn,          // [NN,HH]
                                 bf16*  __restrict__ out_bf,      // or null
                                 float* __restrict__ out_f32,     // or null
                                 int out_ld, int out_off,
                                 int step, int reverse) {
    extern __shared__ char smem[];
    float* sHproj = (float*)smem;                 // [NN][G4] = 128KB

    __shared__ int sSe[E];
    __shared__ int sDe[E];
    __shared__ int sCnt[NN];
    __shared__ int sStart[NN + 1];
    __shared__ int sOrder[E];

    const int tid  = threadIdx.x;
    const int lane = tid & 31;
    const int wave = tid >> 5;

    const int t = reverse ? (T - 1 - step) : step;
    const int* sArr = reverse ? dst : src;
    const int* dArr = reverse ? src : dst;

    if (tid < E) {
        sSe[tid] = sArr[(long)t * E + tid];
        sDe[tid] = dArr[(long)t * E + tid];
    }
    __syncthreads();

    // ---- Phase 1: Hproj = hn @ Whh^T  (M=32, N=1024, K=256) ----
    // 128 16x16 tiles over 32 waves -> 4 tiles per wave, 8 WMMA each.
#pragma unroll
    for (int i = 0; i < 4; ++i) {
        int tt    = wave + 32 * i;       // 0..127
        int mtile = tt & 1;
        int ntile = tt >> 1;
        v8f acc = {};
        const bf16* Abase = hn  + (long)(mtile * 16) * HH;
        const bf16* Bbase = Whh + (long)(ntile * 16) * HH;
#pragma unroll
        for (int k0 = 0; k0 < HH; k0 += 32) {
            v16bf a = load_frag16(Abase + k0, HH, lane);
            v16bf b = load_frag16(Bbase + k0, HH, lane);
            acc = __builtin_amdgcn_wmma_f32_16x16x32_bf16(
                false, a, false, b, (short)0, acc, false, false);
        }
        int m = mtile * 16 + (lane >> 4) * 8;
        int n = ntile * 16 + (lane & 15);
#pragma unroll
        for (int v = 0; v < 8; ++v) sHproj[(m + v) * G4 + n] = acc[v];
    }

    // ---- Phase 2: deterministic per-node edge lists ----
    if (tid < NN) {
        int c = 0;
        for (int e = 0; e < E; ++e) if (sDe[e] == tid) ++c;
        sCnt[tid] = c;
    }
    __syncthreads();
    if (tid == 0) {
        int s = 0;
        for (int n = 0; n < NN; ++n) { sStart[n] = s; s += sCnt[n]; }
        sStart[NN] = s;
    }
    __syncthreads();
    if (tid < NN) {
        int k = sStart[tid];
        for (int e = 0; e < E; ++e) if (sDe[e] == tid) sOrder[k++] = e;
    }
    __syncthreads();

    // ---- Phase 3: LSTM cell + fixed-order segment mean ----
    const long tE = (long)t * E;
    const bool wbf = (out_bf != nullptr);
    float hres[8], cres[8];
#pragma unroll
    for (int p = 0; p < 8; ++p) {
        int pid = tid + p * 1024;        // 8192 (node,h) pairs
        int n = pid >> 8;
        int h = pid & 255;
        int s0 = sStart[n], s1 = sStart[n + 1];
        float hsum = 0.0f, csum = 0.0f;
        for (int idx = s0; idx < s1; ++idx) {
            int e  = sOrder[idx];
            int sn = sSe[e];
            const float* gb = Gin + (tE + e) * (long)G4;
            const float* hp = sHproj + sn * G4;
            float gi = gb[h]       + hp[h];
            float gf = gb[256 + h] + hp[256 + h];
            float gg = gb[512 + h] + hp[512 + h];
            float go = gb[768 + h] + hp[768 + h];
            float cs = cn[sn * HH + h];
            float cc = sigmoidf(gf) * cs + sigmoidf(gi) * tanhf(gg);
            float hh = sigmoidf(go) * tanhf(cc);
            long oi = (tE + e) * (long)out_ld + out_off + h;
            if (wbf) out_bf[oi] = (bf16)hh; else out_f32[oi] = hh;
            hsum += hh; csum += cc;
        }
        int len = s1 - s0;
        float denom = (len > 0) ? (float)len : 1.0f;
        hres[p] = hsum / denom;
        cres[p] = csum / denom;
    }
    __syncthreads();   // all reads of old hn/cn complete before overwrite
#pragma unroll
    for (int p = 0; p < 8; ++p) {
        int pid = tid + p * 1024;
        int n = pid >> 8;
        int h = pid & 255;
        hn[n * HH + h] = (bf16)hres[p];
        cn[n * HH + h] = cres[p];
    }
}

// ---------------------------------------------------------------------------
// Host orchestration
// ---------------------------------------------------------------------------
extern "C" void kernel_launch(void* const* d_in, const int* in_sizes, int n_in,
                              void* d_out, int out_size, void* d_ws, size_t ws_size,
                              hipStream_t stream) {
    const float* x     = (const float*)d_in[0];   // [T,E,64]
    const float* Wih0  = (const float*)d_in[1];   // [2,1024,64]
    const float* Whh0  = (const float*)d_in[2];   // [2,1024,256]
    const float* b0    = (const float*)d_in[3];   // [2,1024]
    const float* Wih1  = (const float*)d_in[4];   // [2,1024,512]
    const float* Whh1  = (const float*)d_in[5];   // [2,1024,256]
    const float* b1    = (const float*)d_in[6];   // [2,1024]
    const int*   src   = (const int*)d_in[7];     // [T,E]
    const int*   dst   = (const int*)d_in[8];     // [T,E]
    float*       out   = (float*)d_out;           // [T,E,512]

    // ---- workspace carve ----
    char* ws = (char*)d_ws;
    size_t off = 0;
    auto alloc = [&](size_t bytes) -> void* {
        off = (off + 255) & ~(size_t)255;
        void* p = ws + off;
        off += bytes;
        return p;
    };
    const long ME = (long)T * E;                          // 65536 rows
    bf16*  Xbf0   = (bf16*) alloc(ME * DIN0 * sizeof(bf16));   // 8 MB
    bf16*  X1bf   = (bf16*) alloc(ME * DIN1 * sizeof(bf16));   // 64 MB
    bf16*  Wih0b  = (bf16*) alloc(2L * G4 * DIN0 * sizeof(bf16));
    bf16*  Whh0b  = (bf16*) alloc(2L * G4 * HH   * sizeof(bf16));
    bf16*  Wih1b  = (bf16*) alloc(2L * G4 * DIN1 * sizeof(bf16));
    bf16*  Whh1b  = (bf16*) alloc(2L * G4 * HH   * sizeof(bf16));
    float* Gin    = (float*)alloc(ME * G4 * sizeof(float));    // 256 MB
    bf16*  hn     = (bf16*) alloc((size_t)NN * HH * sizeof(bf16));
    float* cn     = (float*)alloc((size_t)NN * HH * sizeof(float));
    (void)ws_size; (void)in_sizes; (void)n_in; (void)out_size;

    auto cvt = [&](const float* s, bf16* d, long n) {
        int blocks = (int)((n + 255) / 256);
        if (blocks > 16384) blocks = 16384;
        cvt_bf16_kernel<<<blocks, 256, 0, stream>>>(s, d, n);
    };
    cvt(x,    Xbf0,  ME * DIN0);
    cvt(Wih0, Wih0b, 2L * G4 * DIN0);
    cvt(Whh0, Whh0b, 2L * G4 * HH);
    cvt(Wih1, Wih1b, 2L * G4 * DIN1);
    cvt(Whh1, Whh1b, 2L * G4 * HH);

    auto proj = [&](const bf16* X, const bf16* W, const float* bias, int K) {
        int jobs = (int)(ME / 16) * (G4 / 64);     // 65536 wave-jobs
        proj_gemm_kernel<<<jobs / 8, 256, 0, stream>>>(X, W, bias, Gin,
                                                       (int)ME, G4, K);
    };
    const size_t stepLds = (size_t)NN * G4 * sizeof(float);   // 128 KB

    // ---------- layer 0 ----------
    for (int dir = 0; dir < 2; ++dir) {
        proj(Xbf0, Wih0b + (long)dir * G4 * DIN0, b0 + dir * G4, DIN0);
        init_state_kernel<<<(NN * HH + 255) / 256, 256, 0, stream>>>(hn, cn);
        const bf16* Wr = Whh0b + (long)dir * G4 * HH;
        for (int step = 0; step < T; ++step) {
            lstm_step_kernel<<<1, 1024, stepLds, stream>>>(
                Gin, Wr, src, dst, hn, cn,
                /*out_bf=*/X1bf, /*out_f32=*/nullptr,
                /*out_ld=*/DIN1, /*out_off=*/dir * HH, step, dir);
        }
    }
    // ---------- layer 1 (writes final f32 output) ----------
    for (int dir = 0; dir < 2; ++dir) {
        proj(X1bf, Wih1b + (long)dir * G4 * DIN1, b1 + dir * G4, DIN1);
        init_state_kernel<<<(NN * HH + 255) / 256, 256, 0, stream>>>(hn, cn);
        const bf16* Wr = Whh1b + (long)dir * G4 * HH;
        for (int step = 0; step < T; ++step) {
            lstm_step_kernel<<<1, 1024, stepLds, stream>>>(
                Gin, Wr, src, dst, hn, cn,
                /*out_bf=*/nullptr, /*out_f32=*/out,
                /*out_ld=*/2 * HH, /*out_off=*/dir * HH, step, dir);
        }
    }
}
